// Cross_MultiAttention_80015240724553
// MI455X (gfx1250) — matrix operand order repaired and checked by reading, not compiled, not run on hardware
//
#include <hip/hip_runtime.h>

// ---------------------------------------------------------------------------
// Cross-attention for MI455X (gfx1250).
//  - All GEMM-shaped math via V_WMMA_F32_16X16X32_F16 (f16 in, f32 accum).
//  - One-time fp32->f16 conversion of x/context/weights, so GEMM staging is
//    pure 16B copies -> GLOBAL_LOAD_ASYNC_TO_LDS_B128 (ASYNCcnt) when the
//    toolchain declares the builtin, with a sync v8h fallback.
//  - Double-buffered K-loop (BK=64): 8 WMMAs per barrier pair.
// ---------------------------------------------------------------------------

typedef _Float16 v16h __attribute__((ext_vector_type(16)));
typedef _Float16 v8h  __attribute__((ext_vector_type(8)));
typedef _Float16 h4   __attribute__((ext_vector_type(4)));
typedef float    v8f  __attribute__((ext_vector_type(8)));
typedef float    f4   __attribute__((ext_vector_type(4)));
typedef int      v4i  __attribute__((ext_vector_type(4)));

#define NUM_HEADS 16
#define EMB_DIM   1024
#define IN_CH     512
#define D_CTX     768
#define DEPTH     64
#define BATCH     4
#define NQ        4096
#define SK        256
#define ATT_SCALE 0.125f     /* 64^-0.5 */
#define NEGVAL    (-1.0e9f)

#if defined(__has_builtin)
#  if __has_builtin(__builtin_amdgcn_global_load_async_to_lds_b128) && \
      __has_builtin(__builtin_amdgcn_s_wait_asynccnt)
#    define ASYNC_LDS 1
#  endif
#endif
#ifndef ASYNC_LDS
#  define ASYNC_LDS 0
#endif

__device__ __forceinline__ v8h ld8(const _Float16* p) {
    return *reinterpret_cast<const v8h*>(p);
}
__device__ __forceinline__ v16h cat8(v8h lo, v8h hi) {
    return __builtin_shufflevector(lo, hi, 0,1,2,3,4,5,6,7,8,9,10,11,12,13,14,15);
}
__device__ __forceinline__ v8f wmma_f16(v16h a, v16h b, v8f c) {
    return __builtin_amdgcn_wmma_f32_16x16x32_f16(false, a, false, b, (short)0, c,
                                                  false, false);
}
// 16-byte global -> LDS copy (async on CDNA5 when available)
__device__ __forceinline__ void cp16_lds(_Float16* l, const _Float16* g) {
#if ASYNC_LDS
    __builtin_amdgcn_global_load_async_to_lds_b128(
        (__attribute__((address_space(1))) v4i*)(g),
        (__attribute__((address_space(3))) v4i*)(l), 0, 0);
#else
    *reinterpret_cast<v8h*>(l) = ld8(g);
#endif
}

// ---------------------------------------------------------------------------
// One-time fp32 -> f16 conversion (vectorized x4).
// ---------------------------------------------------------------------------
__global__ __launch_bounds__(256)
void cvt_f32_to_f16(const float* __restrict__ s, _Float16* __restrict__ d, int n4)
{
    int i = blockIdx.x * 256 + threadIdx.x;
    if (i < n4) {
        f4 v = reinterpret_cast<const f4*>(s)[i];
        h4 h; h[0] = (_Float16)v[0]; h[1] = (_Float16)v[1];
              h[2] = (_Float16)v[2]; h[3] = (_Float16)v[3];
        reinterpret_cast<h4*>(d)[i] = h;
    }
}

// ---------------------------------------------------------------------------
// GEMM: C[M,N] = act(X[M,K] @ W[N,K]^T + bias[N]), X/W already f16.
// Block tile 128x64, 256 threads (8 wave32), BK=64, double-buffered LDS,
// async global->LDS staging. Row stride 72 halves (144B) -> conflict-free
// b128 fragment gathers.
// ---------------------------------------------------------------------------
template <bool RELU, typename OT>
__global__ __launch_bounds__(256)
void gemm_wmma(const _Float16* __restrict__ X, const _Float16* __restrict__ W,
               const float* __restrict__ bias, OT* __restrict__ C,
               int M, int Kd, int Nout)
{
    constexpr int BM = 128, BN = 64, BK = 64, ST = BK + 8;  // 72 halves
    __shared__ __align__(16) _Float16 Xs[2][BM * ST];
    __shared__ __align__(16) _Float16 Ws[2][BN * ST];

    const int tid  = threadIdx.x;
    const int wave = tid >> 5;
    const int lane = tid & 31;
    const int m0   = blockIdx.x * BM;
    const int n0   = blockIdx.y * BN;

    const int rA    = lane & 15;
    const int koffA = (lane < 16) ? 0 : 16;
    const int cB    = lane & 15;
    const int koffB = (lane < 16) ? 0 : 16;

    v8f acc[4] = {{0.f,0.f,0.f,0.f,0.f,0.f,0.f,0.f},
                  {0.f,0.f,0.f,0.f,0.f,0.f,0.f,0.f},
                  {0.f,0.f,0.f,0.f,0.f,0.f,0.f,0.f},
                  {0.f,0.f,0.f,0.f,0.f,0.f,0.f,0.f}};

    // stage one K-slab (6 x 16B copies per thread: 4 for X, 2 for W)
    auto stage = [&](int buf, int k0) {
        #pragma unroll
        for (int i = tid; i < (BM * BK) / 8; i += 256) {          // 4 per thread
            int row = i >> 3, cg = (i & 7) * 8;
            cp16_lds(&Xs[buf][row * ST + cg],
                     X + (size_t)(m0 + row) * Kd + k0 + cg);
        }
        #pragma unroll
        for (int i = tid; i < (BN * BK) / 8; i += 256) {          // 2 per thread
            int row = i >> 3, cg = (i & 7) * 8;
            cp16_lds(&Ws[buf][row * ST + cg],
                     W + (size_t)(n0 + row) * Kd + k0 + cg);
        }
    };

    const int nk = Kd / BK;
    stage(0, 0);
    for (int kk = 0; kk < nk; ++kk) {
        const int cur = kk & 1;
        const bool more = (kk + 1) < nk;
        if (more) stage(cur ^ 1, (kk + 1) * BK);
#if ASYNC_LDS
        if (more) __builtin_amdgcn_s_wait_asynccnt(6);   // only next slab in flight
        else      __builtin_amdgcn_s_wait_asynccnt(0);
#endif
        __syncthreads();

        #pragma unroll
        for (int kc = 0; kc < 2; ++kc) {
            const _Float16* xrow = &Xs[cur][(wave * 16 + rA) * ST + kc * 32];
            v16h a = cat8(ld8(xrow + koffA), ld8(xrow + koffA + 8));
            #pragma unroll
            for (int nt = 0; nt < 4; ++nt) {
                const _Float16* wrow = &Ws[cur][(nt * 16 + cB) * ST + kc * 32 + koffB];
                v16h b = cat8(ld8(wrow), ld8(wrow + 8));
                acc[nt] = wmma_f16(a, b, acc[nt]);
            }
        }
        __syncthreads();
    }

    // epilogue: bias (+ReLU), store
    const int rowBase = m0 + wave * 16 + ((lane < 16) ? 0 : 8);
    #pragma unroll
    for (int nt = 0; nt < 4; ++nt) {
        int col = n0 + nt * 16 + (lane & 15);
        float bv = bias[col];
        #pragma unroll
        for (int r = 0; r < 8; ++r) {
            float v = acc[nt][r] + bv;
            if (RELU) v = v > 0.f ? v : 0.f;
            size_t off = (size_t)(rowBase + r) * Nout + col;
            if constexpr (sizeof(OT) == 2) C[off] = (OT)(_Float16)v;
            else                           C[off] = (OT)v;
        }
    }
}

// ---------------------------------------------------------------------------
// Flash-style cross attention. Block = (b,h) x 64 query rows, 4 wave32.
// S in chunks of 32; online softmax; P re-striped via per-wave LDS;
// K chunk staged with async copies, V chunk transposed element-wise.
// ---------------------------------------------------------------------------
__global__ __launch_bounds__(128)
void attn_wmma(const _Float16* __restrict__ Q, const _Float16* __restrict__ K,
               const _Float16* __restrict__ V, const float* __restrict__ pad_mask,
               const float* __restrict__ seq_mask, _Float16* __restrict__ O)
{
    const int bid = blockIdx.x;
    const int nb  = bid & 63;
    const int h   = (bid >> 6) & 15;
    const int b   = bid >> 10;
    const int n0  = nb * 64;
    const int tid = threadIdx.x, wave = tid >> 5, lane = tid & 31;
    const int wrow = n0 + wave * 16;

    __shared__ __align__(16) _Float16 Kc[32 * 72];      // K chunk  [32s][64d]+pad
    __shared__ __align__(16) _Float16 Vt[64 * 40];      // V chunk^T[64d][32s]+pad
    __shared__ __align__(16) _Float16 Ps[4][16 * 40];   // per-wave P tile

    const int rA    = lane & 15;
    const int koffA = (lane < 16) ? 0 : 16;
    const int cB    = lane & 15;
    const int koffB = (lane < 16) ? 0 : 16;

    const _Float16* qrow = Q + ((size_t)b * NQ + wrow + rA) * EMB_DIM + h * DEPTH;
    v16h qa0 = cat8(ld8(qrow + koffA),      ld8(qrow + koffA + 8));
    v16h qa1 = cat8(ld8(qrow + 32 + koffA), ld8(qrow + 32 + koffA + 8));

    const int rowBase = wrow + ((lane < 16) ? 0 : 8);
    float pm[8];
    #pragma unroll
    for (int r = 0; r < 8; ++r) pm[r] = pad_mask[(size_t)b * NQ + rowBase + r];

    v8f o[4] = {{0.f,0.f,0.f,0.f,0.f,0.f,0.f,0.f},
                {0.f,0.f,0.f,0.f,0.f,0.f,0.f,0.f},
                {0.f,0.f,0.f,0.f,0.f,0.f,0.f,0.f},
                {0.f,0.f,0.f,0.f,0.f,0.f,0.f,0.f}};
    float mrow[8], lrow[8];
    #pragma unroll
    for (int r = 0; r < 8; ++r) { mrow[r] = -3.0e38f; lrow[r] = 0.f; }

    _Float16* pw = Ps[wave];

    for (int sc = 0; sc < SK / 32; ++sc) {
        __syncthreads();
        // ---- stage K chunk [32][64] (async, 2 x b128 per thread) ----
        #pragma unroll
        for (int i = tid; i < 256; i += 128) {
            int row = i >> 3, seg = i & 7;
            cp16_lds(&Kc[row * 72 + seg * 8],
                     K + ((size_t)b * SK + sc * 32 + row) * EMB_DIM
                       + h * DEPTH + seg * 8);
        }
        // ---- stage V chunk transposed [64][32] ----
        {
            int sl = tid >> 2, dseg = tid & 3;
            const _Float16* vr = V + ((size_t)b * SK + sc * 32 + sl) * EMB_DIM
                                   + h * DEPTH + dseg * 16;
            v8h v0 = ld8(vr), v1 = ld8(vr + 8);
            #pragma unroll
            for (int e = 0; e < 8; ++e) {
                Vt[(dseg * 16 + e) * 40 + sl]     = v0[e];
                Vt[(dseg * 16 + 8 + e) * 40 + sl] = v1[e];
            }
        }
#if ASYNC_LDS
        __builtin_amdgcn_s_wait_asynccnt(0);
#endif
        __syncthreads();

        // ---- scores: 2 S-tiles x 2 d-chunks ----
        v8f sa[2] = {{0.f,0.f,0.f,0.f,0.f,0.f,0.f,0.f},
                     {0.f,0.f,0.f,0.f,0.f,0.f,0.f,0.f}};
        #pragma unroll
        for (int jl = 0; jl < 2; ++jl) {
            const _Float16* kr = &Kc[(jl * 16 + cB) * 72];
            v16h bk0 = cat8(ld8(kr + koffB),      ld8(kr + koffB + 8));
            sa[jl] = wmma_f16(qa0, bk0, sa[jl]);
            v16h bk1 = cat8(ld8(kr + 32 + koffB), ld8(kr + 32 + koffB + 8));
            sa[jl] = wmma_f16(qa1, bk1, sa[jl]);
        }

        // ---- scale + mask ----
        float smv[2];
        #pragma unroll
        for (int jl = 0; jl < 2; ++jl)
            smv[jl] = seq_mask[(size_t)b * SK + sc * 32 + jl * 16 + (lane & 15)];
        #pragma unroll
        for (int jl = 0; jl < 2; ++jl)
            #pragma unroll
            for (int r = 0; r < 8; ++r) {
                bool ok = (pm[r] != 0.f) && (smv[jl] != 0.f);
                sa[jl][r] = ok ? sa[jl][r] * ATT_SCALE : NEGVAL;
            }

        // ---- online softmax (row stats live in 16-lane halves) ----
        float cmax[8];
        #pragma unroll
        for (int r = 0; r < 8; ++r) cmax[r] = fmaxf(sa[0][r], sa[1][r]);
        #pragma unroll
        for (int off = 1; off < 16; off <<= 1)
            #pragma unroll
            for (int r = 0; r < 8; ++r)
                cmax[r] = fmaxf(cmax[r], __shfl_xor(cmax[r], off, 32));

        float alpha[8], psum[8];
        #pragma unroll
        for (int r = 0; r < 8; ++r) {
            float mn = fmaxf(mrow[r], cmax[r]);
            alpha[r] = __expf(mrow[r] - mn);
            mrow[r]  = mn;
            psum[r]  = 0.f;
        }
        #pragma unroll
        for (int jl = 0; jl < 2; ++jl)
            #pragma unroll
            for (int r = 0; r < 8; ++r) {
                float p = __expf(sa[jl][r] - mrow[r]);
                sa[jl][r] = p;
                psum[r] += p;
            }
        #pragma unroll
        for (int off = 1; off < 16; off <<= 1)
            #pragma unroll
            for (int r = 0; r < 8; ++r)
                psum[r] += __shfl_xor(psum[r], off, 32);
        #pragma unroll
        for (int r = 0; r < 8; ++r) lrow[r] = lrow[r] * alpha[r] + psum[r];
        #pragma unroll
        for (int nt = 0; nt < 4; ++nt)
            #pragma unroll
            for (int r = 0; r < 8; ++r) o[nt][r] *= alpha[r];

        // ---- re-stripe P (C-layout -> A-layout) via per-wave LDS ----
        const int rOff = (lane < 16) ? 0 : 8;
        #pragma unroll
        for (int jl = 0; jl < 2; ++jl)
            #pragma unroll
            for (int r = 0; r < 8; ++r)
                pw[(rOff + r) * 40 + jl * 16 + (lane & 15)] = (_Float16)sa[jl][r];
        const _Float16* prow = pw + rA * 40;   // per-wave DS ops are in-order
        v16h pa = cat8(ld8(prow + koffA), ld8(prow + koffA + 8));

        // ---- P @ V ----
        #pragma unroll
        for (int nt = 0; nt < 4; ++nt) {
            const _Float16* vrow = &Vt[(nt * 16 + cB) * 40 + koffB];
            v16h bv = cat8(ld8(vrow), ld8(vrow + 8));
            o[nt] = wmma_f16(pa, bv, o[nt]);
        }
    }

    // ---- normalize + store f16 ----
    #pragma unroll
    for (int nt = 0; nt < 4; ++nt) {
        int col = h * DEPTH + nt * 16 + (lane & 15);
        #pragma unroll
        for (int r = 0; r < 8; ++r) {
            float v = o[nt][r] / lrow[r];
            O[((size_t)b * NQ + rowBase + r) * EMB_DIM + col] = (_Float16)v;
        }
    }
}

// ---------------------------------------------------------------------------
extern "C" void kernel_launch(void* const* d_in, const int* in_sizes, int n_in,
                              void* d_out, int out_size, void* d_ws, size_t ws_size,
                              hipStream_t stream)
{
    (void)in_sizes; (void)n_in; (void)out_size; (void)ws_size;
    const float* x    = (const float*)d_in[0];
    const float* ctx  = (const float*)d_in[1];
    const float* pad  = (const float*)d_in[2];
    const float* seq  = (const float*)d_in[3];
    const float* Wq   = (const float*)d_in[4];
    const float* Wqb  = (const float*)d_in[5];
    const float* Wk   = (const float*)d_in[6];
    const float* Wkb  = (const float*)d_in[7];
    const float* Wv   = (const float*)d_in[8];
    const float* Wvb  = (const float*)d_in[9];
    const float* P1   = (const float*)d_in[10];
    const float* P1b  = (const float*)d_in[11];
    const float* P2   = (const float*)d_in[12];
    const float* P2b  = (const float*)d_in[13];
    float* out = (float*)d_out;

    const int MQ = BATCH * NQ;   // 16384
    const int MS = BATCH * SK;   // 1024

    char* ws = (char*)d_ws;
    auto carve = [&](size_t elems) {
        _Float16* p = (_Float16*)ws; ws += elems * sizeof(_Float16); return p;
    };
    _Float16* Qb  = carve((size_t)MQ * EMB_DIM);
    _Float16* Kb  = carve((size_t)MS * EMB_DIM);
    _Float16* Vb  = carve((size_t)MS * EMB_DIM);
    _Float16* Ab  = carve((size_t)MQ * EMB_DIM);
    _Float16* Hb  = carve((size_t)MQ * EMB_DIM);
    _Float16* xh  = carve((size_t)MQ * IN_CH);
    _Float16* ch  = carve((size_t)MS * D_CTX);
    _Float16* wqh = carve((size_t)EMB_DIM * IN_CH);
    _Float16* wkh = carve((size_t)EMB_DIM * D_CTX);
    _Float16* wvh = carve((size_t)EMB_DIM * D_CTX);
    _Float16* p1h = carve((size_t)EMB_DIM * EMB_DIM);
    _Float16* p2h = carve((size_t)IN_CH * EMB_DIM);

    auto cvt = [&](const float* s, _Float16* d, size_t n) {
        int n4 = (int)(n / 4);
        cvt_f32_to_f16<<<(n4 + 255) / 256, 256, 0, stream>>>(s, d, n4);
    };
    cvt(x,   xh,  (size_t)MQ * IN_CH);
    cvt(ctx, ch,  (size_t)MS * D_CTX);
    cvt(Wq,  wqh, (size_t)EMB_DIM * IN_CH);
    cvt(Wk,  wkh, (size_t)EMB_DIM * D_CTX);
    cvt(Wv,  wvh, (size_t)EMB_DIM * D_CTX);
    cvt(P1,  p1h, (size_t)EMB_DIM * EMB_DIM);
    cvt(P2,  p2h, (size_t)IN_CH * EMB_DIM);

    dim3 tb(256);
    gemm_wmma<false, _Float16>
        <<<dim3(MQ / 128, EMB_DIM / 64), tb, 0, stream>>>(xh, wqh, Wqb, Qb, MQ, IN_CH, EMB_DIM);
    gemm_wmma<false, _Float16>
        <<<dim3(MS / 128, EMB_DIM / 64), tb, 0, stream>>>(ch, wkh, Wkb, Kb, MS, D_CTX, EMB_DIM);
    gemm_wmma<false, _Float16>
        <<<dim3(MS / 128, EMB_DIM / 64), tb, 0, stream>>>(ch, wvh, Wvb, Vb, MS, D_CTX, EMB_DIM);

    attn_wmma<<<BATCH * NUM_HEADS * (NQ / 64), dim3(128), 0, stream>>>(
        Qb, Kb, Vb, pad, seq, Ab);

    gemm_wmma<true, _Float16>
        <<<dim3(MQ / 128, EMB_DIM / 64), tb, 0, stream>>>(Ab, p1h, P1b, Hb, MQ, EMB_DIM, EMB_DIM);
    gemm_wmma<false, float>
        <<<dim3(MQ / 128, IN_CH / 64), tb, 0, stream>>>(Hb, p2h, P2b, out, MQ, EMB_DIM, IN_CH);
}
